// MultiheadAttention_3702261809698
// MI455X (gfx1250) — matrix operand order repaired and checked by reading, not compiled, not run on hardware
//
#include <hip/hip_runtime.h>
#include <hip/hip_bf16.h>
#include <cstdint>
#include <cstddef>

typedef __bf16 bf16;
typedef __attribute__((ext_vector_type(16))) __bf16 v16bf;
typedef __attribute__((ext_vector_type(8)))  __bf16 v8bf;
typedef __attribute__((ext_vector_type(4)))  __bf16 v4bf;
typedef __attribute__((ext_vector_type(8)))  float  v8f;

union FragBF { v16bf v; v8bf h[2]; };

#define WMMA_BF16(A,B,C) \
  __builtin_amdgcn_wmma_f32_16x16x32_bf16(false,(A),false,(B),(short)0,(C),false,false)

// Problem constants
#define NB 2
#define NT 4096
#define NE 1024
#define NH 16
#define ND 64
#define NM (NB*NT)   // 8192 rows

// ---------------------------------------------------------------------------
// gfx1250 async global->LDS copy (32 bytes per lane, 2 instructions).
// The instruction offset is added to both the LDS and the global address.
// ---------------------------------------------------------------------------
__device__ __forceinline__ uint32_t lds_addr_of(const void* p) {
  // generic pointers to LDS carry the wave-relative LDS offset in addr[31:0]
  return (uint32_t)(uintptr_t)p;
}

__device__ __forceinline__ void async_copy_32B(uint32_t lds_addr, const bf16* gptr) {
  asm volatile(
      "global_load_async_to_lds_b128 %0, %1, off\n\t"
      "global_load_async_to_lds_b128 %0, %1, off offset:16"
      :: "v"(lds_addr), "v"(gptr) : "memory");
}

__device__ __forceinline__ void async_wait0() {
  asm volatile("s_wait_asynccnt 0" ::: "memory");
}

// ---------------------------------------------------------------------------
// f32 -> bf16 converter (4 elements / thread)
// ---------------------------------------------------------------------------
__global__ void cvt_f32_bf16(const float* __restrict__ in, bf16* __restrict__ out, int n4) {
  int i = blockIdx.x * blockDim.x + threadIdx.x;
  if (i >= n4) return;
  float4 f = ((const float4*)in)[i];
  v4bf r;
  r[0] = (bf16)f.x; r[1] = (bf16)f.y; r[2] = (bf16)f.z; r[3] = (bf16)f.w;
  ((v4bf*)out)[i] = r;
}

// ---------------------------------------------------------------------------
// QKV projection: Y = X @ W^T   (X: 8192x1024 bf16, W: 1024x1024 bf16 (n,k))
// blockIdx.z selects Wq/Wk/Wv. Q,K stored (B,H,T,D); V stored (B,H,D,T).
// Block: 256 thr (8 waves). Macro tile 128x128, wave tile 64x32, K-step 32.
// Async double-buffered LDS staging.
// ---------------------------------------------------------------------------
__global__ __launch_bounds__(256)
void qkv_gemm(const bf16* __restrict__ X,
              const bf16* __restrict__ Wq, const bf16* __restrict__ Wk,
              const bf16* __restrict__ Wv,
              bf16* __restrict__ Qb, bf16* __restrict__ Kb, bf16* __restrict__ Vb) {
  constexpr int LDA = 40;                     // 80B row stride (16B aligned)
  __shared__ __align__(16) bf16 sA[2][128 * LDA];
  __shared__ __align__(16) bf16 sB[2][128 * LDA];

  const int which = blockIdx.z;
  const bf16* W = (which == 0) ? Wq : (which == 1) ? Wk : Wv;

  const int m0   = blockIdx.x * 128;
  const int n0   = blockIdx.y * 128;
  const int tid  = threadIdx.x;
  const int lane = tid & 31;
  const int wave = tid >> 5;
  const int lh   = lane >> 4;                 // half-wave id
  const int l16  = lane & 15;
  const int wm   = wave >> 2;                 // 0..1
  const int wn   = wave & 3;                  // 0..3

  v8f acc[4][2];
  for (int mi = 0; mi < 4; ++mi)
    for (int ni = 0; ni < 2; ++ni) { v8f z = {}; acc[mi][ni] = z; }

  const int rowL = tid >> 1;                  // 0..127
  const int segL = (tid & 1) * 16;            // 0 / 16
  const bf16* gA = X + (size_t)(m0 + rowL) * NE + segL;
  const bf16* gB = W + (size_t)(n0 + rowL) * NE + segL;
  const uint32_t ldsOff = (uint32_t)(rowL * LDA + segL);

  // Prologue: fetch k-tile 0 into buffer 0
  async_copy_32B(lds_addr_of(&sA[0][ldsOff]), gA);
  async_copy_32B(lds_addr_of(&sB[0][ldsOff]), gB);
  async_wait0();
  __syncthreads();

  int buf = 0;
  for (int k0 = 0; k0 < NE; k0 += 32) {
    if (k0 + 32 < NE) {                       // prefetch next k-tile
      async_copy_32B(lds_addr_of(&sA[buf ^ 1][ldsOff]), gA + k0 + 32);
      async_copy_32B(lds_addr_of(&sB[buf ^ 1][ldsOff]), gB + k0 + 32);
      __builtin_prefetch(gA + k0 + 96, 0, 1);
      __builtin_prefetch(gB + k0 + 96, 0, 1);
    }

    FragBF bfr[2];
    for (int ni = 0; ni < 2; ++ni) {          // B operand: interleaved halves
      const bf16* p = &sB[buf][(wn * 32 + ni * 16 + l16) * LDA];
      bfr[ni].h[0] = *(const v8bf*)(p + lh * 8);
      bfr[ni].h[1] = *(const v8bf*)(p + 16 + lh * 8);
    }
    for (int mi = 0; mi < 4; ++mi) {          // A operand: interleaved halves
      FragBF af;
      const bf16* p = &sA[buf][(wm * 64 + mi * 16 + l16) * LDA];
      af.h[0] = *(const v8bf*)(p + lh * 8);
      af.h[1] = *(const v8bf*)(p + 16 + lh * 8);
      for (int ni = 0; ni < 2; ++ni)
        acc[mi][ni] = WMMA_BF16(af.v, bfr[ni].v, acc[mi][ni]);
    }

    async_wait0();
    __syncthreads();
    buf ^= 1;
  }

  // Epilogue: C layout row = i + 8*lh, col = l16
  for (int mi = 0; mi < 4; ++mi)
    for (int ni = 0; ni < 2; ++ni)
      for (int i = 0; i < 8; ++i) {
        int gm = m0 + wm * 64 + mi * 16 + i + lh * 8;
        int gn = n0 + wn * 32 + ni * 16 + l16;
        int b  = gm >> 12;                    // / 4096
        int t  = gm & (NT - 1);
        int h  = gn >> 6;
        int d  = gn & (ND - 1);
        bf16 v = (bf16)acc[mi][ni][i];
        if (which == 2)
          Vb[(((size_t)b * NH + h) * ND + d) * NT + t] = v;   // (B,H,D,T)
        else {
          bf16* dst = (which == 0) ? Qb : Kb;
          dst[(((size_t)b * NH + h) * NT + t) * ND + d] = v;  // (B,H,T,D)
        }
      }
}

// ---------------------------------------------------------------------------
// Flash attention: per (b,h), 128 query rows / block, 8 waves x 16 rows.
// kv tiles of 64 async-staged into double-buffered LDS; online softmax.
// ---------------------------------------------------------------------------
__global__ __launch_bounds__(256)
void flash_attn(const bf16* __restrict__ Qb, const bf16* __restrict__ Kb,
                const bf16* __restrict__ Vb, bf16* __restrict__ Ob) {
  constexpr int LDK = 72;                     // 144B row stride
  __shared__ __align__(16) bf16 sK[2][64 * LDK];
  __shared__ __align__(16) bf16 sV[2][64 * LDK];
  __shared__ __align__(16) bf16 sP[8 * 16 * LDK];

  const int bh    = blockIdx.y;               // 0..31  (b*16 + h)
  const int q0blk = blockIdx.x * 128;
  const int tid   = threadIdx.x;
  const int lane  = tid & 31;
  const int wave  = tid >> 5;
  const int lh    = lane >> 4;
  const int l16   = lane & 15;
  const int qbase = q0blk + wave * 16;

  const bf16* Qh = Qb + (size_t)bh * NT * ND;
  const bf16* Kh = Kb + (size_t)bh * NT * ND;
  const bf16* Vh = Vb + (size_t)bh * ND * NT; // row = d, col = t

  // Q fragments held in registers for the whole kv sweep
  FragBF qf[2];
  {
    const bf16* qp = Qh + (size_t)(qbase + l16) * ND;
    for (int kk = 0; kk < 2; ++kk) {
      qf[kk].h[0] = *(const v8bf*)(qp + kk * 32 + lh * 8);
      qf[kk].h[1] = *(const v8bf*)(qp + kk * 32 + 16 + lh * 8);
    }
  }

  v8f o[4];
  for (int j = 0; j < 4; ++j) { v8f z = {}; o[j] = z; }
  float mrow[8], lrow[8];
  for (int i = 0; i < 8; ++i) { mrow[i] = -3.0e38f; lrow[i] = 0.0f; }

  const float scale = 0.125f;                 // 1/sqrt(64)
  const float LOG2E = 1.4426950408889634f;

  const int kvend = q0blk + 128;
  const int ldrow = tid >> 2;                 // 0..63
  const int ldseg = (tid & 3) * 16;           // 0,16,32,48
  bf16* sPw = &sP[wave * 16 * LDK];

  const bf16* gK = Kh + (size_t)ldrow * ND + ldseg;   // + kv*ND per tile
  const bf16* gV = Vh + (size_t)ldrow * NT + ldseg;   // + kv   per tile
  const uint32_t kOff = (uint32_t)(ldrow * LDK + ldseg);

  // Prologue: fetch kv tile 0 into buffer 0
  async_copy_32B(lds_addr_of(&sK[0][kOff]), gK);
  async_copy_32B(lds_addr_of(&sV[0][kOff]), gV);
  async_wait0();
  __syncthreads();

  int buf = 0;
  for (int kv0 = 0; kv0 < kvend; kv0 += 64) {
    if (kv0 + 64 < kvend) {                   // prefetch next kv tile
      async_copy_32B(lds_addr_of(&sK[buf ^ 1][kOff]), gK + (size_t)(kv0 + 64) * ND);
      async_copy_32B(lds_addr_of(&sV[buf ^ 1][kOff]), gV + (kv0 + 64));
    }

    // S = (Q K^T): 4 tiles of 16 kv columns
    v8f s[4];
    for (int j = 0; j < 4; ++j) {
      v8f z = {}; s[j] = z;
      for (int kk = 0; kk < 2; ++kk) {
        FragBF kf;                            // B operand: lane = kv column
        const bf16* p = &sK[buf][(j * 16 + l16) * LDK + kk * 32];
        kf.h[0] = *(const v8bf*)(p + lh * 8);
        kf.h[1] = *(const v8bf*)(p + 16 + lh * 8);
        s[j] = WMMA_BF16(qf[kk].v, kf.v, s[j]);
      }
    }

    // Online softmax (per-row stats; rows live in 16-lane halves)
    float fs[8];
    for (int i = 0; i < 8; ++i) {
      int grow = qbase + i + lh * 8;
      float mx = -3.0e38f;
      for (int j = 0; j < 4; ++j) {
        int gcol = kv0 + j * 16 + l16;
        float v = s[j][i] * scale;
        v = (gcol <= grow) ? v : -3.0e38f;    // causal mask
        s[j][i] = v;
        mx = fmaxf(mx, v);
      }
      mx = fmaxf(mx, __shfl_xor(mx, 1));
      mx = fmaxf(mx, __shfl_xor(mx, 2));
      mx = fmaxf(mx, __shfl_xor(mx, 4));
      mx = fmaxf(mx, __shfl_xor(mx, 8));
      float mnew = fmaxf(mrow[i], mx);
      fs[i] = exp2f((mrow[i] - mnew) * LOG2E);
      float rs = 0.0f;
      for (int j = 0; j < 4; ++j) {
        float p = exp2f((s[j][i] - mnew) * LOG2E);
        s[j][i] = p;
        rs += p;
      }
      rs += __shfl_xor(rs, 1);
      rs += __shfl_xor(rs, 2);
      rs += __shfl_xor(rs, 4);
      rs += __shfl_xor(rs, 8);
      lrow[i] = lrow[i] * fs[i] + rs;
      mrow[i] = mnew;
    }

    // Rescale O, stage P (C layout -> A layout) through per-wave LDS strip
    for (int j = 0; j < 4; ++j)
      for (int i = 0; i < 8; ++i)
        o[j][i] *= fs[i];
    for (int j = 0; j < 4; ++j)
      for (int i = 0; i < 8; ++i)
        sPw[(i + lh * 8) * LDK + j * 16 + l16] = (bf16)s[j][i];

    // O += P V
    FragBF pf[2];
    for (int kk = 0; kk < 2; ++kk) {
      const bf16* p = &sPw[l16 * LDK];
      pf[kk].h[0] = *(const v8bf*)(p + kk * 32 + lh * 8);
      pf[kk].h[1] = *(const v8bf*)(p + kk * 32 + 16 + lh * 8);
    }
    for (int j2 = 0; j2 < 4; ++j2)
      for (int kk = 0; kk < 2; ++kk) {
        FragBF vf;                            // B operand: lane = d column
        const bf16* p = &sV[buf][(j2 * 16 + l16) * LDK + kk * 32];
        vf.h[0] = *(const v8bf*)(p + lh * 8);
        vf.h[1] = *(const v8bf*)(p + 16 + lh * 8);
        o[j2] = WMMA_BF16(pf[kk].v, vf.v, o[j2]);
      }

    async_wait0();
    __syncthreads();
    buf ^= 1;
  }

  // Normalize and store O as (B,T,E) bf16
  const int b = bh >> 4, h = bh & 15;
  for (int j2 = 0; j2 < 4; ++j2)
    for (int i = 0; i < 8; ++i) {
      int gr = qbase + i + lh * 8;
      int gc = h * ND + j2 * 16 + l16;
      float v = o[j2][i] / lrow[i];
      Ob[((size_t)b * NT + gr) * NE + gc] = (bf16)v;
    }
}

// ---------------------------------------------------------------------------
// Output projection: out = O @ Wo^T + bo  (fp32 out), async double-buffered
// ---------------------------------------------------------------------------
__global__ __launch_bounds__(256)
void out_proj(const bf16* __restrict__ A, const bf16* __restrict__ W,
              const float* __restrict__ bias, float* __restrict__ out) {
  constexpr int LDA = 40;
  __shared__ __align__(16) bf16 sA[2][128 * LDA];
  __shared__ __align__(16) bf16 sB[2][128 * LDA];

  const int m0   = blockIdx.x * 128;
  const int n0   = blockIdx.y * 128;
  const int tid  = threadIdx.x;
  const int lane = tid & 31;
  const int wave = tid >> 5;
  const int lh   = lane >> 4;
  const int l16  = lane & 15;
  const int wm   = wave >> 2;
  const int wn   = wave & 3;

  v8f acc[4][2];
  for (int mi = 0; mi < 4; ++mi)
    for (int ni = 0; ni < 2; ++ni) { v8f z = {}; acc[mi][ni] = z; }

  const int rowL = tid >> 1;
  const int segL = (tid & 1) * 16;
  const bf16* gA = A + (size_t)(m0 + rowL) * NE + segL;
  const bf16* gB = W + (size_t)(n0 + rowL) * NE + segL;
  const uint32_t ldsOff = (uint32_t)(rowL * LDA + segL);

  async_copy_32B(lds_addr_of(&sA[0][ldsOff]), gA);
  async_copy_32B(lds_addr_of(&sB[0][ldsOff]), gB);
  async_wait0();
  __syncthreads();

  int buf = 0;
  for (int k0 = 0; k0 < NE; k0 += 32) {
    if (k0 + 32 < NE) {
      async_copy_32B(lds_addr_of(&sA[buf ^ 1][ldsOff]), gA + k0 + 32);
      async_copy_32B(lds_addr_of(&sB[buf ^ 1][ldsOff]), gB + k0 + 32);
      __builtin_prefetch(gA + k0 + 96, 0, 1);
      __builtin_prefetch(gB + k0 + 96, 0, 1);
    }

    FragBF bfr[2];
    for (int ni = 0; ni < 2; ++ni) {
      const bf16* p = &sB[buf][(wn * 32 + ni * 16 + l16) * LDA];
      bfr[ni].h[0] = *(const v8bf*)(p + lh * 8);
      bfr[ni].h[1] = *(const v8bf*)(p + 16 + lh * 8);
    }
    for (int mi = 0; mi < 4; ++mi) {
      FragBF af;
      const bf16* p = &sA[buf][(wm * 64 + mi * 16 + l16) * LDA];
      af.h[0] = *(const v8bf*)(p + lh * 8);
      af.h[1] = *(const v8bf*)(p + 16 + lh * 8);
      for (int ni = 0; ni < 2; ++ni)
        acc[mi][ni] = WMMA_BF16(af.v, bfr[ni].v, acc[mi][ni]);
    }

    async_wait0();
    __syncthreads();
    buf ^= 1;
  }

  for (int mi = 0; mi < 4; ++mi)
    for (int ni = 0; ni < 2; ++ni)
      for (int i = 0; i < 8; ++i) {
        int gm = m0 + wm * 64 + mi * 16 + i + lh * 8;
        int gn = n0 + wn * 32 + ni * 16 + l16;
        out[(size_t)gm * NE + gn] = acc[mi][ni][i] + bias[gn];
      }
}

// ---------------------------------------------------------------------------
extern "C" void kernel_launch(void* const* d_in, const int* in_sizes, int n_in,
                              void* d_out, int out_size, void* d_ws, size_t ws_size,
                              hipStream_t stream) {
  const float* x  = (const float*)d_in[0];
  const float* Wq = (const float*)d_in[1];
  const float* Wk = (const float*)d_in[2];
  const float* Wv = (const float*)d_in[3];
  const float* Wo = (const float*)d_in[4];
  const float* bo = (const float*)d_in[5];
  float* out = (float*)d_out;

  char* ws = (char*)d_ws;
  size_t off = 0;
  const size_t szX = (size_t)NM * NE * sizeof(bf16);     // 16 MB
  const size_t szW = (size_t)NE * NE * sizeof(bf16);     // 2 MB
  bf16* xb  = (bf16*)(ws + off); off += szX;
  bf16* wqb = (bf16*)(ws + off); off += szW;
  bf16* wkb = (bf16*)(ws + off); off += szW;
  bf16* wvb = (bf16*)(ws + off); off += szW;
  bf16* wob = (bf16*)(ws + off); off += szW;
  bf16* Qb  = (bf16*)(ws + off); off += szX;             // (B,H,T,D)
  bf16* Kb  = (bf16*)(ws + off); off += szX;             // (B,H,T,D)
  bf16* Vb  = (bf16*)(ws + off); off += szX;             // (B,H,D,T)
  bf16* Ob  = (bf16*)(ws + off); off += szX;             // (B,T,E)

  const int nX4 = NM * NE / 4;
  const int nW4 = NE * NE / 4;
  cvt_f32_bf16<<<nX4 / 256, 256, 0, stream>>>(x, xb, nX4);
  cvt_f32_bf16<<<nW4 / 256, 256, 0, stream>>>(Wq, wqb, nW4);
  cvt_f32_bf16<<<nW4 / 256, 256, 0, stream>>>(Wk, wkb, nW4);
  cvt_f32_bf16<<<nW4 / 256, 256, 0, stream>>>(Wv, wvb, nW4);
  cvt_f32_bf16<<<nW4 / 256, 256, 0, stream>>>(Wo, wob, nW4);

  qkv_gemm<<<dim3(NM / 128, NE / 128, 3), 256, 0, stream>>>(
      xb, wqb, wkb, wvb, Qb, Kb, Vb);

  flash_attn<<<dim3(NT / 128, NB * NH), 256, 0, stream>>>(Qb, Kb, Vb, Ob);

  out_proj<<<dim3(NM / 128, NE / 128), 256, 0, stream>>>(Ob, wob, bo, out);
}
